// MultiHeadRot_86706799772407
// MI455X (gfx1250) — matrix operand order repaired and checked by reading, not compiled, not run on hardware
//
#include <hip/hip_runtime.h>
#include <hip/hip_bf16.h>

typedef __bf16 bf16;
typedef __attribute__((ext_vector_type(16))) __bf16 v16bf;
typedef __attribute__((ext_vector_type(8)))  float  v8f;

#define B_  4
#define S_  2048
#define D_  1024
#define H_  16
#define HD_ 64

static __device__ __forceinline__ v8f wmma_bf16(v16bf a, v16bf b, v8f c) {
  // D = A(16x32 bf16) * B(32x16 bf16) + C(16x16 f32)
  return __builtin_amdgcn_wmma_f32_16x16x32_bf16(false, a, false, b, (short)0, c, false, false);
}

// ---- Fragment loaders (CDNA5 ISA 7.12.2 16-bit layouts) --------------------
// A 16x32 (MxK), row-major, row stride ld (elements).
static __device__ __forceinline__ v16bf load_a(const bf16* p, int ld, int lane) {
  int row = lane & 15;
  int kb  = (lane >> 4) << 3;   // 0 or 8
  union { v16bf v; uint4 q[2]; } u;
  u.q[0] = *reinterpret_cast<const uint4*>(p + (size_t)row * ld + kb);
  u.q[1] = *reinterpret_cast<const uint4*>(p + (size_t)row * ld + kb + 16);
  return u.v;
}

// B 32x16 (KxN) with memory holding N rows contiguous along K (B^T row-major).
static __device__ __forceinline__ v16bf load_bt(const bf16* p, int ld, int lane) {
  int n  = lane & 15;
  int kb = (lane >> 4) << 3;    // 0 or 8
  union { v16bf v; uint4 q[2]; } u;
  const bf16* r = p + (size_t)n * ld + kb;
  u.q[0] = *reinterpret_cast<const uint4*>(r);
  u.q[1] = *reinterpret_cast<const uint4*>(r + 16);
  return u.v;
}

// ---- f32 -> bf16 convert ----------------------------------------------------
__global__ void cvt_f32_bf16(const float* __restrict__ in, bf16* __restrict__ out, long n) {
  long i = (long)blockIdx.x * blockDim.x + threadIdx.x;
  if (i < n) out[i] = (bf16)in[i];
}

// ---- GEMM: C[M,N] = A[M,K] * Bnk[N,K]^T  (both K-contiguous) ---------------
// One wave computes a 32(M) x 64(N) tile: 2 A-frags x 4 B-frags, 8 accumulators.
template <typename OutT>
__global__ void gemm_xwT(const bf16* __restrict__ A, const bf16* __restrict__ Bm,
                         OutT* __restrict__ C, int M, int N, int K) {
  int lane = threadIdx.x & 31;
  int wave = (int)((blockIdx.x * blockDim.x + threadIdx.x) >> 5);
  int ntiles = N >> 6;
  int tm = wave / ntiles;
  int tn = wave - tm * ntiles;
  if (tm * 32 >= M) return;

  const bf16* A0 = A + (size_t)tm * 32 * K;
  const bf16* A1 = A0 + (size_t)16 * K;
  v8f acc[2][4] = {};
  for (int k = 0; k < K; k += 32) {
    v16bf a0 = load_a(A0 + k, K, lane);
    v16bf a1 = load_a(A1 + k, K, lane);
    if (k + 64 < K) {  // gfx1250 global_prefetch_b8 on the next A K-slab
      __builtin_prefetch(A0 + (size_t)(lane & 15) * K + k + 64, 0, 1);
      __builtin_prefetch(A1 + (size_t)(lane & 15) * K + k + 64, 0, 1);
    }
    const bf16* Bk = Bm + (size_t)(tn * 64) * K + k;
    v16bf b0 = load_bt(Bk,                  K, lane);
    v16bf b1 = load_bt(Bk + (size_t)16 * K, K, lane);
    v16bf b2 = load_bt(Bk + (size_t)32 * K, K, lane);
    v16bf b3 = load_bt(Bk + (size_t)48 * K, K, lane);
    acc[0][0] = wmma_bf16(a0, b0, acc[0][0]);
    acc[1][0] = wmma_bf16(a1, b0, acc[1][0]);
    acc[0][1] = wmma_bf16(a0, b1, acc[0][1]);
    acc[1][1] = wmma_bf16(a1, b1, acc[1][1]);
    acc[0][2] = wmma_bf16(a0, b2, acc[0][2]);
    acc[1][2] = wmma_bf16(a1, b2, acc[1][2]);
    acc[0][3] = wmma_bf16(a0, b3, acc[0][3]);
    acc[1][3] = wmma_bf16(a1, b3, acc[1][3]);
  }
  int rbase = (lane >> 4) << 3;
  int col   = lane & 15;
#pragma unroll
  for (int mi = 0; mi < 2; mi++) {
#pragma unroll
    for (int j = 0; j < 4; j++) {
      size_t cb = (size_t)(tm * 32 + mi * 16 + rbase) * N + tn * 64 + j * 16 + col;
#pragma unroll
      for (int r = 0; r < 8; r++)
        C[cb + (size_t)r * N] = (OutT)acc[mi][j][r];
    }
  }
}

// ---- RoPE + head split; folds 1/sqrt(64) into Q; V stored transposed -------
__global__ void rope_split(const bf16* __restrict__ qkv, bf16* __restrict__ Q,
                           bf16* __restrict__ Kd, bf16* __restrict__ Vt) {
  long tid = (long)blockIdx.x * blockDim.x + threadIdx.x;
  long total = (long)B_ * H_ * S_ * 32;
  if (tid >= total) return;
  int  i  = (int)(tid & 31);
  long t  = tid >> 5;
  int  s  = (int)(t % S_);
  long t2 = t / S_;
  int  h  = (int)(t2 % H_);
  int  b  = (int)(t2 / H_);

  size_t row = ((size_t)b * S_ + s) * (3 * D_);
  int e = h * HD_ + 2 * i;
  float q0 = (float)qkv[row + e];
  float q1 = (float)qkv[row + e + 1];
  float k0 = (float)qkv[row + D_ + e];
  float k1 = (float)qkv[row + D_ + e + 1];
  float v0 = (float)qkv[row + 2 * D_ + e];
  float v1 = (float)qkv[row + 2 * D_ + e + 1];

  float inv = __expf(-((float)(2 * i) / (float)HD_) * 9.210340371976184f); // ln(1e4)
  float ang = (float)s * inv;
  float sn, c;
  __sincosf(ang, &sn, &c);

  float q0r = q0 * c - q1 * sn, q1r = q1 * c + q0 * sn;
  float k0r = k0 * c - k1 * sn, k1r = k1 * c + k0 * sn;

  size_t hb  = (size_t)(b * H_ + h);
  size_t qki = (hb * S_ + s) * HD_ + 2 * i;
  Q[qki]     = (bf16)(q0r * 0.125f);
  Q[qki + 1] = (bf16)(q1r * 0.125f);
  Kd[qki]     = (bf16)k0r;
  Kd[qki + 1] = (bf16)k1r;
  size_t vi = (hb * HD_ + 2 * i) * S_ + s;    // V transposed [BH, 64, S]
  Vt[vi]      = (bf16)v0;
  Vt[vi + S_] = (bf16)v1;
}

// ---- Flash attention: one wave = 16 query rows, KV tiles of 32 -------------
// 16-row waves keep peak pressure ~150 VGPRs (no spills, 2x occupancy of a
// 32-row wave). K/V fragments are named and hoisted so they issue as one deep
// load clause with staggered s_wait_loadcnt into the WMMAs.
#define SC_LD 36   // f32 score row stride (16B-aligned, bank-skewed)
#define P_LD  40   // bf16 P row stride (16B-aligned, bank-skewed)

__global__ void flash_attn(const bf16* __restrict__ Qb, const bf16* __restrict__ Kb,
                           const bf16* __restrict__ Vt, bf16* __restrict__ Ob) {
  __shared__ __align__(16) float sc[8][16 * SC_LD];
  __shared__ __align__(16) bf16  ps[8][16 * P_LD];
  __shared__ __align__(16) float bc[8][16];
  int lane = threadIdx.x & 31;
  int w    = threadIdx.x >> 5;
  int gwave = blockIdx.x * 8 + w;
  int bh = gwave >> 7;          // / (S/16)
  int qt = gwave & 127;
  if (bh >= B_ * H_) return;

  int qbase = qt * 16;
  const bf16* Q = Qb + ((size_t)bh * S_ + qbase) * HD_;
  const bf16* K = Kb + (size_t)bh * S_ * HD_;
  const bf16* V = Vt + (size_t)bh * HD_ * S_;

  v16bf qf0 = load_a(Q,      HD_, lane);
  v16bf qf1 = load_a(Q + 32, HD_, lane);

  v8f o0 = {}, o1 = {}, o2 = {}, o3 = {};
  int half  = lane >> 4;
  int col   = lane & 15;
  int myrow = lane & 15;        // row owned in the row-major softmax pass
  int cb    = half << 4;        // column half owned in that pass
  float m = -1e30f, l = 0.f;    // per-row stats (replicated in lane^16)
  int rowg = qbase + myrow;
  int ktmax = (qbase + 15) >> 5;

  for (int kt = 0; kt <= ktmax; kt++) {
    int kvb = kt * 32;
    // ---- QK^T: 4 named K-fragments -> one deep load clause, 4 WMMAs ----
    v16bf kb0 = load_bt(K + (size_t)kvb * HD_,             HD_, lane);
    v16bf kb1 = load_bt(K + (size_t)kvb * HD_ + 32,        HD_, lane);
    v16bf kb2 = load_bt(K + (size_t)(kvb + 16) * HD_,      HD_, lane);
    v16bf kb3 = load_bt(K + (size_t)(kvb + 16) * HD_ + 32, HD_, lane);
    v8f s0 = {}, s1 = {};
    s0 = wmma_bf16(qf0, kb0, s0);
    s1 = wmma_bf16(qf0, kb2, s1);
    s0 = wmma_bf16(qf1, kb1, s0);
    s1 = wmma_bf16(qf1, kb3, s1);

    // C-fragment -> row-major LDS scores
#pragma unroll
    for (int r = 0; r < 8; r++) {
      int ra = (half << 3) + r;
      sc[w][ra * SC_LD + col]      = s0[r];
      sc[w][ra * SC_LD + col + 16] = s1[r];
    }
    asm volatile("s_wait_dscnt 0" ::: "memory");

    // ---- row-major softmax: lane owns (myrow, cols cb..cb+15) ----
    float e[16];
    {
      const float* rp = &sc[w][myrow * SC_LD + cb];
      float4 t0 = *(const float4*)(rp);
      float4 t1 = *(const float4*)(rp + 4);
      float4 t2 = *(const float4*)(rp + 8);
      float4 t3 = *(const float4*)(rp + 12);
      e[0]=t0.x; e[1]=t0.y; e[2]=t0.z;  e[3]=t0.w;
      e[4]=t1.x; e[5]=t1.y; e[6]=t1.z;  e[7]=t1.w;
      e[8]=t2.x; e[9]=t2.y; e[10]=t2.z; e[11]=t2.w;
      e[12]=t3.x;e[13]=t3.y;e[14]=t3.z; e[15]=t3.w;
    }
#pragma unroll
    for (int j = 0; j < 16; j++)
      if (kvb + cb + j > rowg) e[j] = -1e30f;     // causal mask
    float mx = e[0];
#pragma unroll
    for (int j = 1; j < 16; j++) mx = fmaxf(mx, e[j]);
    mx = fmaxf(mx, __shfl_xor(mx, 16, 32));       // combine the two row halves
    float mnew = fmaxf(m, mx);
    float scl  = __expf(m - mnew);
    float sum  = 0.f;
    union { bf16 h[16]; uint4 q[2]; } pb;
#pragma unroll
    for (int j = 0; j < 16; j++) {
      float p = __expf(e[j] - mnew);
      sum += p;
      pb.h[j] = (bf16)p;
    }
    sum += __shfl_xor(sum, 16, 32);
    l = l * scl + sum;
    m = mnew;
    *(uint4*)&ps[w][myrow * P_LD + cb]     = pb.q[0];
    *(uint4*)&ps[w][myrow * P_LD + cb + 8] = pb.q[1];
    if (lane < 16) bc[w][lane] = scl;             // per-row rescale broadcast
    asm volatile("s_wait_dscnt 0" ::: "memory");

    // ---- rescale O accumulators (C layout rows: half*8 + r) ----
    {
      float4 sa = *(const float4*)&bc[w][half * 8];
      float4 sb = *(const float4*)&bc[w][half * 8 + 4];
      float sr[8] = {sa.x, sa.y, sa.z, sa.w, sb.x, sb.y, sb.z, sb.w};
#pragma unroll
      for (int r = 0; r < 8; r++) {
        o0[r] *= sr[r]; o1[r] *= sr[r]; o2[r] *= sr[r]; o3[r] *= sr[r];
      }
    }

    // ---- P @ V: 4 named V-fragments -> one deep load clause, 4 WMMAs ----
    v16bf pf  = load_a(&ps[w][0], P_LD, lane);
    v16bf vb0 = load_bt(V +           kvb, S_, lane);
    v16bf vb1 = load_bt(V + 16 * S_ + kvb, S_, lane);
    v16bf vb2 = load_bt(V + 32 * S_ + kvb, S_, lane);
    v16bf vb3 = load_bt(V + 48 * S_ + kvb, S_, lane);
    o0 = wmma_bf16(pf, vb0, o0);
    o1 = wmma_bf16(pf, vb1, o1);
    o2 = wmma_bf16(pf, vb2, o2);
    o3 = wmma_bf16(pf, vb3, o3);
  }

  // ---- final 1/l normalization, broadcast through LDS ----
  if (lane < 16) bc[w][lane] = 1.f / l;
  asm volatile("s_wait_dscnt 0" ::: "memory");
  float4 sa = *(const float4*)&bc[w][half * 8];
  float4 sb = *(const float4*)&bc[w][half * 8 + 4];
  float sr[8] = {sa.x, sa.y, sa.z, sa.w, sb.x, sb.y, sb.z, sb.w};

  int b_ = bh >> 4, h_ = bh & 15;
#pragma unroll
  for (int r = 0; r < 8; r++) {
    size_t q    = (size_t)qbase + (half << 3) + r;
    size_t base = ((size_t)b_ * S_ + q) * D_ + h_ * HD_ + col;
    Ob[base]      = (bf16)(o0[r] * sr[r]);
    Ob[base + 16] = (bf16)(o1[r] * sr[r]);
    Ob[base + 32] = (bf16)(o2[r] * sr[r]);
    Ob[base + 48] = (bf16)(o3[r] * sr[r]);
  }
}

extern "C" void kernel_launch(void* const* d_in, const int* in_sizes, int n_in,
                              void* d_out, int out_size, void* d_ws, size_t ws_size,
                              hipStream_t stream) {
  (void)in_sizes; (void)n_in; (void)out_size; (void)ws_size;
  const float* x     = (const float*)d_in[0];
  const float* w_qkv = (const float*)d_in[1];
  const float* w_out = (const float*)d_in[2];
  float* out = (float*)d_out;

  char* ws = (char*)d_ws;
  const size_t NX   = (size_t)B_ * S_ * D_;
  const size_t NWQ  = (size_t)3 * D_ * D_;
  const size_t NWO  = (size_t)D_ * D_;
  const size_t NQKV = (size_t)B_ * S_ * 3 * D_;
  bf16* Xb   = (bf16*)ws; ws += NX   * 2;
  bf16* Wqb  = (bf16*)ws; ws += NWQ  * 2;
  bf16* Wob  = (bf16*)ws; ws += NWO  * 2;
  bf16* QKVb = (bf16*)ws; ws += NQKV * 2;
  bf16* Qb   = (bf16*)ws; ws += NX   * 2;
  bf16* Kb   = (bf16*)ws; ws += NX   * 2;
  bf16* Vtb  = (bf16*)ws; ws += NX   * 2;
  bf16* Ob   = (bf16*)ws; ws += NX   * 2;   // ~143 MB total

  cvt_f32_bf16<<<(int)((NX  + 255) / 256), 256, 0, stream>>>(x,     Xb,  (long)NX);
  cvt_f32_bf16<<<(int)((NWQ + 255) / 256), 256, 0, stream>>>(w_qkv, Wqb, (long)NWQ);
  cvt_f32_bf16<<<(int)((NWO + 255) / 256), 256, 0, stream>>>(w_out, Wob, (long)NWO);

  { // qkv = x @ w_qkv^T : M=8192, N=3072, K=1024 (32x64 tile per wave)
    int M = B_ * S_, N = 3 * D_, K = D_;
    int waves = (M / 32) * (N / 64);
    gemm_xwT<bf16><<<waves * 32 / 256, 256, 0, stream>>>(Xb, Wqb, QKVb, M, N, K);
  }
  { // RoPE + split into head-major Q/K and transposed V
    long total = (long)B_ * H_ * S_ * 32;
    rope_split<<<(int)((total + 255) / 256), 256, 0, stream>>>(QKVb, Qb, Kb, Vtb);
  }
  { // causal flash attention: 16 q-rows per wave
    int waves = B_ * H_ * (S_ / 16);          // 8192 waves
    flash_attn<<<waves / 8, 256, 0, stream>>>(Qb, Kb, Vtb, Ob);
  }
  { // out = O @ w_out^T : M=8192, N=1024, K=1024
    int M = B_ * S_, N = D_, K = D_;
    int waves = (M / 32) * (N / 64);
    gemm_xwT<float><<<waves * 32 / 256, 256, 0, stream>>>(Ob, Wob, out, M, N, K);
  }
}
